// SS2D_68582037782956
// MI455X (gfx1250) — compile-verified, matrix operand order checked by reading, not run on hardware
//
#include <hip/hip_runtime.h>
#include <hip/hip_bf16.h>
#include <math.h>

// ---------------------------------------------------------------------------
// SS2D (VMamba-style 2D selective scan) for MI455X / gfx1250 (wave32, WMMA).
//
//   B=2, H=W=64, d_model=96, d_inner=192, K=4 directions, N=16 state,
//   R=6 dt-rank, L = 32*32 = 1024 per-direction sequence length.
//
// v3 changes vs v2:
//   * GEMM K-loops fully unrolled -> all WMMAs visible to the scheduler,
//     loads hoisted across slabs, no per-slab branch overhead.
//   * Token index decomposition via shifts/masks (all dims are pow2).
//   Kept from v2: 16x32 tiles per wave (A reuse, 2 accumulators),
//   v_rcp/v_log fast activations, double-buffered async-LDS scan staging.
// ---------------------------------------------------------------------------

typedef __attribute__((ext_vector_type(16))) _Float16 v16h;
typedef __attribute__((ext_vector_type(8)))  float    v8f;

constexpr int Bsz  = 2;
constexpr int Hh   = 64;
constexpr int Ww   = 64;
constexpr int DM   = 96;    // d_model
constexpr int DI   = 192;   // d_inner
constexpr int KK   = 4;     // directions
constexpr int NN   = 16;    // d_state
constexpr int RR   = 6;     // dt rank
constexpr int H2   = 32;
constexpr int W2   = 32;
constexpr int LL   = H2 * W2;       // 1024
constexpr int PROJ = RR + 2 * NN;   // 38
constexpr int MTOK = Bsz * Hh * Ww; // 8192 tokens

// K-index inside a 16x32 (A) / 32x16 (B) f16 WMMA operand for element e of
// this lane's v16h, given lane-half (lane>>4).  Per CDNA5 ISA 7.12.2.
__device__ __forceinline__ int wmma_kmap(int half, int e) {
  return (e < 8) ? (half * 8 + e) : (16 + half * 8 + (e - 8));
}

__device__ __forceinline__ v16h load_f16_operand(const float* __restrict__ p,
                                                 long kstride, int kBase,
                                                 int half, bool valid) {
  v16h v;
#pragma unroll
  for (int e = 0; e < 16; ++e) {
    int k = kBase + wmma_kmap(half, e);
    float f = valid ? p[(long)k * kstride] : 0.0f;
    v[e] = (_Float16)f;
  }
  return v;
}

// silu via v_exp_f32 + v_rcp_f32 (no IEEE divide chain)
__device__ __forceinline__ float silu_f(float x) {
  return x * __builtin_amdgcn_rcpf(1.0f + __expf(-x));
}

// (h,w) parity -> (direction k, sequence position l); matches EfficientScan
// scatter and EfficientMerge gather.
__device__ __forceinline__ void map_kl(int h, int w, int& k, int& l) {
  int he = h & 1, we = w & 1;
  if (!he && !we)      { k = 0; l = (h >> 1) * W2 + (w >> 1); }
  else if (he && !we)  { k = 1; l = (w >> 1) * H2 + (h >> 1); }
  else if (!he)        { k = 2; l = (h >> 1) * W2 + (w >> 1); }
  else                 { k = 3; l = (w >> 1) * H2 + (h >> 1); }
}

// CDNA5 async global->LDS copy, 16B per lane (VGLOBAL GLOBAL_LOAD_ASYNC_TO_LDS_B128).
__device__ __forceinline__ void async_load_b128(unsigned lds_off, const float* gaddr) {
  asm volatile("global_load_async_to_lds_b128 %0, %1, off"
               :: "v"(lds_off), "v"(gaddr) : "memory");
}

// ---------------------------------------------------------------------------
// 1) in_proj GEMM: (8192 x 384) = x(8192x96) @ W(384x96)^T.
//    One wave per 16x32 output tile (A reused across both 16-col halves).
// ---------------------------------------------------------------------------
__global__ void __launch_bounds__(256)
k_inproj(const float* __restrict__ x, const float* __restrict__ w,
         float* __restrict__ xc, float* __restrict__ zb) {
  const int NP = (2 * DI) / 32;                      // 12 column-pairs
  int wave = blockIdx.x * (blockDim.x >> 5) + (threadIdx.x >> 5);
  int lane = threadIdx.x & 31;
  int mt = wave / NP, np = wave % NP;
  int half = lane >> 4, lo = lane & 15;

  const float* aRow  = x + (long)(mt * 16 + lo) * DM;
  const float* bRow0 = w + (long)(np * 32 + lo) * DM;
  const float* bRow1 = bRow0 + 16 * DM;

  v8f acc0 = {}, acc1 = {};
#pragma unroll
  for (int kb = 0; kb < DM; kb += 32) {
    v16h a  = load_f16_operand(aRow, 1, kb, half, true);
    v16h b0 = load_f16_operand(bRow0, 1, kb, half, true);
    v16h b1 = load_f16_operand(bRow1, 1, kb, half, true);
    acc0 = __builtin_amdgcn_wmma_f32_16x16x32_f16(false, a, false, b0,
                                                  (short)0, acc0, false, false);
    acc1 = __builtin_amdgcn_wmma_f32_16x16x32_f16(false, a, false, b1,
                                                  (short)0, acc1, false, false);
  }

  int n0 = np * 32 + lo;        // pair regions are 32-wide, DI boundary aligned
#pragma unroll
  for (int r = 0; r < 8; ++r) {
    int m = mt * 16 + r + 8 * half;
    int bb = m >> 12;                    // / (Hh*Ww) = 4096
    int hw = m & 4095;
    int h = hw >> 6, wq = hw & 63;       // / Ww, % Ww
    if (n0 < DI) {                       // uniform per wave
      xc[(((long)bb * DI + n0) * Hh + h) * Ww + wq]        = acc0[r];
      xc[(((long)bb * DI + n0 + 16) * Hh + h) * Ww + wq]   = acc1[r];
    } else {
      zb[(long)m * DI + (n0 - DI)]      = silu_f(acc0[r]);
      zb[(long)m * DI + (n0 + 16 - DI)] = silu_f(acc1[r]);
    }
  }
}

// ---------------------------------------------------------------------------
// 2) depthwise 3x3 conv + bias + silu, scatter into scan layout xs (B,K,C,L)
// ---------------------------------------------------------------------------
__global__ void __launch_bounds__(256)
k_conv_scan(const float* __restrict__ xc, const float* __restrict__ cw,
            const float* __restrict__ cb, float* __restrict__ xs) {
  int idx = blockIdx.x * blockDim.x + threadIdx.x;   // < 2^21, fits int
  int wq = idx & 63;
  int h  = (idx >> 6) & 63;
  int c  = (idx >> 12) % DI;
  int b  = idx / (4096 * DI);

  const float* base = xc + ((long)b * DI + c) * Hh * Ww;
  const float* wgt  = cw + (long)c * 9;
  float acc = cb[c];
#pragma unroll
  for (int dy = 0; dy < 3; ++dy) {
    int hh = h + dy - 1;
    if ((unsigned)hh >= (unsigned)Hh) continue;
#pragma unroll
    for (int dx = 0; dx < 3; ++dx) {
      int ww2 = wq + dx - 1;
      if ((unsigned)ww2 >= (unsigned)Ww) continue;
      acc = fmaf(base[hh * Ww + ww2], wgt[dy * 3 + dx], acc);
    }
  }
  float s = silu_f(acc);
  int k, l;
  map_kl(h, wq, k, l);
  xs[(((long)b * KK + k) * DI + c) * LL + l] = s;
}

// ---------------------------------------------------------------------------
// 3) x_proj: per (b,k) GEMM  x_dbl(38 x 1024) = Wk(38x192) @ xs_bk(192x1024)
//    One wave per 16x32 tile (A = weight rows, reused).
// ---------------------------------------------------------------------------
__global__ void __launch_bounds__(256)
k_xproj(const float* __restrict__ xs, const float* __restrict__ xw,
        float* __restrict__ xdbl) {
  const int MT = 3, NTP = LL / 32;                   // 3 x 32 tile-pairs per (b,k)
  int wave = blockIdx.x * (blockDim.x >> 5) + (threadIdx.x >> 5);
  int lane = threadIdx.x & 31;
  int bk  = wave / (MT * NTP);
  int rem = wave % (MT * NTP);
  int mt = rem / NTP, np = rem % NTP;
  int k = bk % KK;
  int half = lane >> 4, lo = lane & 15;

  int mRow = mt * 16 + lo;                                      // c' in [0,38)
  const float* aRow  = xw + ((long)k * PROJ + mRow) * DI;
  const float* bCol0 = xs + ((long)bk * DI) * LL + (np * 32 + lo);   // + d*LL
  const float* bCol1 = bCol0 + 16;

  v8f acc0 = {}, acc1 = {};
#pragma unroll
  for (int kb = 0; kb < DI; kb += 32) {
    v16h a  = load_f16_operand(aRow, 1, kb, half, mRow < PROJ);
    v16h b0 = load_f16_operand(bCol0, LL, kb, half, true);
    v16h b1 = load_f16_operand(bCol1, LL, kb, half, true);
    acc0 = __builtin_amdgcn_wmma_f32_16x16x32_f16(false, a, false, b0,
                                                  (short)0, acc0, false, false);
    acc1 = __builtin_amdgcn_wmma_f32_16x16x32_f16(false, a, false, b1,
                                                  (short)0, acc1, false, false);
  }

  int l0 = np * 32 + lo;
#pragma unroll
  for (int r = 0; r < 8; ++r) {
    int m = mt * 16 + r + 8 * half;
    if (m < PROJ) {
      xdbl[((long)bk * PROJ + m) * LL + l0]      = acc0[r];
      xdbl[((long)bk * PROJ + m) * LL + l0 + 16] = acc1[r];
    }
  }
}

// ---------------------------------------------------------------------------
// 4) delta = softplus(dts @ dt_proj_w^T + dt_proj_b)   (R=6 -> plain VALU)
// ---------------------------------------------------------------------------
__global__ void __launch_bounds__(256)
k_delta(const float* __restrict__ xdbl, const float* __restrict__ dtw,
        const float* __restrict__ dtb, float* __restrict__ delta) {
  int idx = blockIdx.x * blockDim.x + threadIdx.x;   // < 2^21
  int l  = idx & (LL - 1);
  int d  = (idx >> 10) % DI;
  int bk = idx / (LL * DI);
  int k  = bk % KK;

  const float* row = xdbl + (long)bk * PROJ * LL + l;   // r stride = LL
  const float* wr  = dtw + ((long)k * DI + d) * RR;
  float acc = dtb[k * DI + d];
#pragma unroll
  for (int r = 0; r < RR; ++r) acc = fmaf(row[(long)r * LL], wr[r], acc);
  delta[idx] = (acc > 20.0f) ? acc : __logf(1.0f + __expf(acc));
}

// ---------------------------------------------------------------------------
// 5) selective scan: 1 block per (b,k), 1 thread per channel c, h[16] in
//    registers.  B_l/C_l for 16 steps at a time are DMA'd into LDS with
//    double-buffered GLOBAL_LOAD_ASYNC_TO_LDS_B128 (waves 0/1 issue, all 6
//    waves consume), so the serial recurrence only touches LDS + registers.
// ---------------------------------------------------------------------------
__global__ void __launch_bounds__(DI)
k_scan(const float* __restrict__ xs, const float* __restrict__ xdbl,
       const float* __restrict__ delta, const float* __restrict__ A_logs,
       const float* __restrict__ Ds, float* __restrict__ ys) {
  int bk = blockIdx.x;
  int k  = bk % KK;
  int c  = threadIdx.x;

  __shared__ __align__(16) float sB[2][NN * 16];   // [buf][n*16 + j]
  __shared__ __align__(16) float sC[2][NN * 16];

  float a[NN], h[NN];
  const float* al = A_logs + ((long)k * DI + c) * NN;
#pragma unroll
  for (int n = 0; n < NN; ++n) { a[n] = -__expf(al[n]); h[n] = 0.0f; }
  float Dv = Ds[k * DI + c];

  const float* ub = xs    + ((long)bk * DI + c) * LL;
  const float* db = delta + ((long)bk * DI + c) * LL;
  const float* Bb = xdbl  + ((long)bk * PROJ + RR) * LL;
  const float* Cb = xdbl  + ((long)bk * PROJ + RR + NN) * LL;
  float* yb = ys + ((long)bk * DI + c) * LL;

  const int NCH = LL / 16;                         // 64 chunks of 16 steps
  int wv = threadIdx.x >> 5;
  int ln = threadIdx.x & 31;

  // Issue async copy of chunk `ch` (16 steps of all 16 B or C rows) into buf.
  auto issue_chunk = [&](int ch, int buf) {
    if (wv > 1) return;                            // waves 0 (B) and 1 (C) only
    const float* src = (wv == 0) ? Bb : Cb;
    float* dst = (wv == 0) ? sB[buf] : sC[buf];
    int row = ln >> 2, q = ln & 3;                 // lane -> (row 0-7, 16B quarter)
    const float* g0 = src + (long)row * LL + ch * 16 + q * 4;
    unsigned d0 = (unsigned)(unsigned long long)(dst + row * 16 + q * 4);
    async_load_b128(d0, g0);                       // rows 0-7
    async_load_b128(d0 + 8 * 16 * 4, g0 + 8 * LL); // rows 8-15
  };

  issue_chunk(0, 0);
  for (int ch = 0; ch < NCH; ++ch) {
    int cur = ch & 1;
    bool more = (ch + 1) < NCH;
    if (more) issue_chunk(ch + 1, cur ^ 1);        // prefetch next chunk
    if (threadIdx.x < 64) {                        // uniform per wave
      if (more) asm volatile("s_wait_asynccnt 2" ::: "memory");  // cur chunk done
      else      asm volatile("s_wait_asynccnt 0" ::: "memory");
    }
    __syncthreads();                               // data visible to all waves

    int l0 = ch * 16;
#pragma unroll 4
    for (int j = 0; j < 16; ++j) {
      int l = l0 + j;
      float dl = db[l];
      float ul = ub[l];
      float du = dl * ul;
      float y  = 0.0f;
#pragma unroll
      for (int n = 0; n < NN; ++n) {
        h[n] = fmaf(__expf(dl * a[n]), h[n], sB[cur][n * 16 + j] * du);
        y    = fmaf(h[n], sC[cur][n * 16 + j], y);
      }
      yb[l] = fmaf(Dv, ul, y);
    }
    __syncthreads();                               // release buf before overwrite
  }
}

// ---------------------------------------------------------------------------
// 6) merge gather + LayerNorm over DI channels + z gating -> ybar (token-major)
// ---------------------------------------------------------------------------
__global__ void __launch_bounds__(DI)
k_merge_ln(const float* __restrict__ ys, const float* __restrict__ zb,
           const float* __restrict__ gamma, const float* __restrict__ beta,
           float* __restrict__ ybar) {
  int m = blockIdx.x;            // token
  int c = threadIdx.x;           // channel
  int b = m >> 12;
  int hw = m & 4095;
  int h = hw >> 6, wq = hw & 63;
  int k, l;
  map_kl(h, wq, k, l);

  float yv = ys[(((long)b * KK + k) * DI + c) * LL + l];

  __shared__ float red[256];
  red[c] = yv;
  if (c < 64) red[DI + c] = 0.0f;
  __syncthreads();
#pragma unroll
  for (int s = 128; s > 0; s >>= 1) {
    if (c < s) red[c] += red[c + s];
    __syncthreads();
  }
  float mu = red[0] * (1.0f / DI);
  __syncthreads();

  float dv = yv - mu;
  red[c] = dv * dv;
  if (c < 64) red[DI + c] = 0.0f;
  __syncthreads();
#pragma unroll
  for (int s = 128; s > 0; s >>= 1) {
    if (c < s) red[c] += red[c + s];
    __syncthreads();
  }
  float var = red[0] * (1.0f / DI);
  float rs  = rsqrtf(var + 1e-5f);

  ybar[(long)m * DI + c] = (dv * rs * gamma[c] + beta[c]) * zb[(long)m * DI + c];
}

// ---------------------------------------------------------------------------
// 7) out_proj GEMM: out(8192x96) = ybar(8192x192) @ Wout(96x192)^T
//    One wave per 16x32 tile.
// ---------------------------------------------------------------------------
__global__ void __launch_bounds__(256)
k_outproj(const float* __restrict__ ybar, const float* __restrict__ ow,
          float* __restrict__ out) {
  const int NP = DM / 32;                            // 3 column-pairs
  int wave = blockIdx.x * (blockDim.x >> 5) + (threadIdx.x >> 5);
  int lane = threadIdx.x & 31;
  int mt = wave / NP, np = wave % NP;
  int half = lane >> 4, lo = lane & 15;

  const float* aRow  = ybar + (long)(mt * 16 + lo) * DI;
  const float* bRow0 = ow   + (long)(np * 32 + lo) * DI;
  const float* bRow1 = bRow0 + 16 * DI;

  v8f acc0 = {}, acc1 = {};
#pragma unroll
  for (int kb = 0; kb < DI; kb += 32) {
    v16h a  = load_f16_operand(aRow, 1, kb, half, true);
    v16h b0 = load_f16_operand(bRow0, 1, kb, half, true);
    v16h b1 = load_f16_operand(bRow1, 1, kb, half, true);
    acc0 = __builtin_amdgcn_wmma_f32_16x16x32_f16(false, a, false, b0,
                                                  (short)0, acc0, false, false);
    acc1 = __builtin_amdgcn_wmma_f32_16x16x32_f16(false, a, false, b1,
                                                  (short)0, acc1, false, false);
  }

  int n0 = np * 32 + lo;
#pragma unroll
  for (int r = 0; r < 8; ++r) {
    int m = mt * 16 + r + 8 * half;
    out[(long)m * DM + n0]      = acc0[r];
    out[(long)m * DM + n0 + 16] = acc1[r];
  }
}

// ---------------------------------------------------------------------------
extern "C" void kernel_launch(void* const* d_in, const int* in_sizes, int n_in,
                              void* d_out, int out_size, void* d_ws, size_t ws_size,
                              hipStream_t stream) {
  const float* x        = (const float*)d_in[0];
  const float* in_w     = (const float*)d_in[1];
  const float* conv_w   = (const float*)d_in[2];
  const float* conv_b   = (const float*)d_in[3];
  const float* xproj_w  = (const float*)d_in[4];
  const float* dtproj_w = (const float*)d_in[5];
  const float* dtproj_b = (const float*)d_in[6];
  const float* A_logs   = (const float*)d_in[7];
  const float* Ds       = (const float*)d_in[8];
  const float* ln_g     = (const float*)d_in[9];
  const float* ln_b     = (const float*)d_in[10];
  const float* out_w    = (const float*)d_in[11];
  float* out = (float*)d_out;

  // Workspace carve-up (floats).  Total ~37.2 MiB.
  const size_t XC_N = (size_t)Bsz * DI * Hh * Ww;   // conv input, NCHW
  const size_t Z_N  = (size_t)MTOK * DI;            // silu(z), token-major
  const size_t XS_N = (size_t)Bsz * KK * DI * LL;   // scan input u
  const size_t XD_N = (size_t)Bsz * KK * PROJ * LL; // dts|Bs|Cs
  const size_t DL_N = XS_N;                         // delta
  const size_t YS_N = XS_N;                         // scan output
  const size_t YB_N = Z_N;                          // post-LN gated activations

  float* p    = (float*)d_ws;
  float* xc   = p;             p += XC_N;
  float* zb   = p;             p += Z_N;
  float* xs   = p;             p += XS_N;
  float* xdbl = p;             p += XD_N;
  float* dl   = p;             p += DL_N;
  float* ysb  = p;             p += YS_N;
  float* ybar = p;             p += YB_N;
  (void)ws_size; (void)in_sizes; (void)n_in; (void)out_size;

  // 1) in_proj: 512*12 wave-tiles, 8 waves/block -> 768 blocks (exact).
  k_inproj<<<(MTOK / 16) * ((2 * DI) / 32) / 8, 256, 0, stream>>>(x, in_w, xc, zb);

  // 2) conv + scatter: 1,572,864 threads.
  k_conv_scan<<<(int)(XC_N / 256), 256, 0, stream>>>(xc, conv_w, conv_b, xs);

  // 3) x_proj: 8*(3*32) = 768 wave-tiles -> 96 blocks (exact).
  k_xproj<<<Bsz * KK * 3 * (LL / 32) / 8, 256, 0, stream>>>(xs, xproj_w, xdbl);

  // 4) dt-proj + softplus.
  k_delta<<<(int)(DL_N / 256), 256, 0, stream>>>(xdbl, dtproj_w, dtproj_b, dl);

  // 5) selective scan: 8 sequences, 192 channels each, async-LDS pipelined.
  k_scan<<<Bsz * KK, DI, 0, stream>>>(xs, xdbl, dl, A_logs, Ds, ysb);

  // 6) merge + LN + gate: one block per token.
  k_merge_ln<<<MTOK, DI, 0, stream>>>(ysb, zb, ln_g, ln_b, ybar);

  // 7) out_proj: 512*3 = 1536 wave-tiles -> 192 blocks (exact).
  k_outproj<<<(MTOK / 16) * (DM / 32) / 8, 256, 0, stream>>>(ybar, out_w, out);
}
